// DGA_70540542869893
// MI455X (gfx1250) — matrix operand order repaired and checked by reading, not compiled, not run on hardware
//
#include <hip/hip_runtime.h>
#include <hip/hip_bf16.h>

// ---------------------------------------------------------------------------
// MI455X / gfx1250 implementation.
// Shapes: B=4, C=64, H=W=256, heads=8, ch=8, HW=65536.
// Memory-bound (~2GB traffic @ 23.3TB/s). WMMA f16-in/f32-acc for the three
// real GEMMs (qkv conv1x1 rgb+depth, fused projection). Tensor Data Mover
// (tensor_load_to_lds + s_wait_tensorcnt) stages the GEMM x-tiles into LDS.
// Fragment-major LDS layouts so WMMA operand loads are ds_load_b128.
// ---------------------------------------------------------------------------

typedef __attribute__((ext_vector_type(16))) _Float16 v16h;
typedef __attribute__((ext_vector_type(8)))  _Float16 v8h;
typedef __attribute__((ext_vector_type(2)))  _Float16 v2h;
typedef __attribute__((ext_vector_type(8)))  float    v8f;
typedef __attribute__((ext_vector_type(4)))  unsigned v4u;
typedef __attribute__((ext_vector_type(4)))  int      v4i;
typedef __attribute__((ext_vector_type(8)))  int      v8i;

#define HWFULL 65536
#define NB 4

#if __has_builtin(__builtin_amdgcn_tensor_load_to_lds)
#define HAVE_TDM 1
#endif

// db3 filters (from reference)
__constant__ float c_dec_lo[6] = { 0.035226291882100656f, -0.08544127388224149f, -0.13501102001039084f,
                                   0.4598775021193313f,    0.8068915093133388f,   0.3326705529509569f };
__constant__ float c_dec_hi[6] = {-0.3326705529509569f,    0.8068915093133388f,  -0.4598775021193313f,
                                  -0.13501102001039084f,   0.08544127388224149f,  0.035226291882100656f };
__constant__ float c_rec_lo[6] = { 0.3326705529509569f,    0.8068915093133388f,   0.4598775021193313f,
                                  -0.13501102001039084f,  -0.08544127388224149f,  0.035226291882100656f };
__constant__ float c_rec_hi[6] = { 0.035226291882100656f,  0.08544127388224149f, -0.13501102001039084f,
                                  -0.4598775021193313f,    0.8068915093133388f,  -0.3326705529509569f };

// ---------------- WMMA helpers (A and B both use contiguous k = 16*(lane/16) + j) ----------------

__device__ inline v8f wmma_f16f32(v16h a, v16h b, v8f c) {
    return __builtin_amdgcn_wmma_f32_16x16x32_f16(
        /*neg_a=*/false, a, /*neg_b=*/false, b,
        /*c_mod=*/(short)0, c, /*reuse_a=*/false, /*reuse_b=*/false);
}

// A-matrix 16x32 f16 from row-major LDS Wl[m*ldw + k].
// lanes 0-15: M=lane, e[0..15]=K kc+0..15; lanes 16-31: +16.
__device__ inline v16h frag_A_rowmajor(const _Float16* Wl, int ldw, int mbase, int kc, int lane) {
    int m  = mbase + (lane & 15);
    int kb = kc + ((lane >= 16) ? 16 : 0);
    const _Float16* p = Wl + m * ldw + kb;
    v8h lo = *(const v8h*)p;            // 16B contiguous -> ds_load_b128
    v8h hi = *(const v8h*)(p + 8);
    return __builtin_shufflevector(lo, hi, 0,1,2,3,4,5,6,7,8,9,10,11,12,13,14,15);
}

// B-matrix 32x16 f16 from fragment-major LDS: Xf[((kc/32)*32 + lane)*16 + j].
// lane l: N=l&15, K = kc + (l>=16?16:0) + j  (j=0..15).
__device__ inline v16h frag_B_fragmajor(const _Float16* Xf, int kc, int lane) {
    return *(const v16h*)(Xf + (((kc >> 5) * 32 + lane) << 4));   // 32B contiguous
}

// ---------------- Tensor Data Mover: 2-D f32 tile -> LDS ---------------------
// D# per CDNA5 ISA ch.8: group0 = {count/flags, lds_addr, global_addr, type=2},
// group1 = {data_size, tensor dims, tile dims, dim0 stride}. 2-D only; groups
// 2/3 zero. EXEC is ignored; call from exactly one wave, then wait TENSORcnt.
__device__ inline void tdm_load_2d_f32(unsigned lds_off, unsigned long long gaddr,
                                       unsigned tile_d0, unsigned tile_d1,
                                       unsigned tensor_d0, unsigned tensor_d1,
                                       unsigned long long stride0_elems) {
#ifdef HAVE_TDM
    v4u g0;
    g0[0] = 1u;                                                  // count=1 (valid user D#)
    g0[1] = lds_off;                                             // lds_addr (bytes)
    g0[2] = (unsigned)(gaddr & 0xFFFFFFFFull);                   // global_addr[31:0]
    g0[3] = (unsigned)((gaddr >> 32) & 0x01FFFFFFull) | 0x80000000u; // [56:32] | type=2
    v8i g1;
    g1[0] = 0x00020000;                                          // data_size=2 (4B), wg_mask=0
    g1[1] = (int)((tensor_d0 & 0xFFFFu) << 16);                  // atomic_addr=0 | tensor_dim0[15:0]
    g1[2] = (int)((tensor_d0 >> 16) | ((tensor_d1 & 0xFFFFu) << 16));
    g1[3] = (int)((tensor_d1 >> 16) | (tile_d0 << 16));          // | tile_dim0
    g1[4] = (int)(tile_d1 & 0xFFFFu);                            // tile_dim1, tile_dim2=0
    g1[5] = (int)(stride0_elems & 0xFFFFFFFFull);                // tensor_dim0_stride[31:0]
    g1[6] = (int)((stride0_elems >> 32) & 0xFFFFull);            // stride[47:32], dim1_stride=0
    g1[7] = 0;
    v4i g2 = {0, 0, 0, 0};
    v4i g3 = {0, 0, 0, 0};
#if __clang_major__ >= 23
    v8i g4 = {0, 0, 0, 0, 0, 0, 0, 0};
    __builtin_amdgcn_tensor_load_to_lds(g0, g1, g2, g3, g4, 0);
#else
    __builtin_amdgcn_tensor_load_to_lds(g0, g1, g2, g3, 0);
#endif
#else
    (void)lds_off; (void)gaddr; (void)tile_d0; (void)tile_d1;
    (void)tensor_d0; (void)tensor_d1; (void)stride0_elems;
#endif
}

__device__ inline void wait_tensorcnt0() {
#if __has_builtin(__builtin_amdgcn_s_wait_tensorcnt)
    __builtin_amdgcn_s_wait_tensorcnt(0);
#else
    asm volatile("s_wait_tensorcnt 0x0" ::: "memory");
#endif
}

// ---------------------------- wavelet kernels -------------------------------

__global__ void dwt_row_kernel(const float* __restrict__ x,
                               float* __restrict__ lo, float* __restrict__ hi, int total) {
    int idx = blockIdx.x * blockDim.x + threadIdx.x;
    if (idx >= total) return;
    int ow = idx % 130;
    int y  = (idx / 130) % 256;
    int bc = idx / (130 * 256);
    const float* row = x + ((size_t)bc * 256 + y) * 256;
    float al = 0.f, ah = 0.f;
#pragma unroll
    for (int t = 0; t < 6; ++t) {
        int xi = 2 * ow + t - 4;
        if (xi >= 0 && xi < 256) {
            float v = row[xi];
            al += c_rec_lo[t] * v;
            ah += c_rec_hi[t] * v;
        }
    }
    lo[idx] = al; hi[idx] = ah;
}

__global__ void dwt_col_kernel(const float* __restrict__ in,
                               float* __restrict__ lo, float* __restrict__ hi, int total) {
    int idx = blockIdx.x * blockDim.x + threadIdx.x;
    if (idx >= total) return;
    int xcol = idx % 130;
    int oy   = (idx / 130) % 130;
    int bc   = idx / (130 * 130);
    float al = 0.f, ah = 0.f;
#pragma unroll
    for (int t = 0; t < 6; ++t) {
        int yi = 2 * oy + t - 4;
        if (yi >= 0 && yi < 256) {
            float v = in[((size_t)bc * 256 + yi) * 130 + xcol];
            al += c_rec_lo[t] * v;
            ah += c_rec_hi[t] * v;
        }
    }
    lo[idx] = al; hi[idx] = ah;
}

__global__ void dw3_kernel(const float* __restrict__ in, const float* __restrict__ w,
                           float* __restrict__ out, int C, int H, int W, int total) {
    int idx = blockIdx.x * blockDim.x + threadIdx.x;
    if (idx >= total) return;
    int x  = idx % W;
    int y  = (idx / W) % H;
    int bc = idx / (W * H);
    int c  = bc % C;
    const float* wp = w + c * 9;
    const float* ip = in + (size_t)bc * H * W;
    float acc = 0.f;
#pragma unroll
    for (int dy = -1; dy <= 1; ++dy) {
        int yy = y + dy;
        if (yy < 0 || yy >= H) continue;
#pragma unroll
        for (int dx = -1; dx <= 1; ++dx) {
            int xx = x + dx;
            if (xx < 0 || xx >= W) continue;
            acc += wp[(dy + 1) * 3 + (dx + 1)] * ip[yy * W + xx];
        }
    }
    out[idx] = acc;
}

__global__ void idwt_col_kernel(const float* __restrict__ a, const float* __restrict__ b,
                                float* __restrict__ out, int total) {
    int idx = blockIdx.x * blockDim.x + threadIdx.x;
    if (idx >= total) return;
    int x  = idx % 130;
    int y  = (idx / 130) % 256;
    int bc = idx / (130 * 256);
    float acc = 0.f;
#pragma unroll
    for (int t = 0; t < 6; ++t) {
        int u = y - 1 + t;                 // zero-upsampled (len 259), pad 1
        if (u >= 0 && u <= 258 && (u & 1) == 0) {
            int iy = u >> 1;
            size_t o = ((size_t)bc * 130 + iy) * 130 + x;
            acc += c_dec_lo[t] * a[o] + c_dec_hi[t] * b[o];
        }
    }
    out[idx] = acc;
}

__global__ void idwt_row_kernel(const float* __restrict__ loW, const float* __restrict__ hiW,
                                float* __restrict__ q, int total) {
    int idx = blockIdx.x * blockDim.x + threadIdx.x;
    if (idx >= total) return;
    int x  = idx % 256;
    int y  = (idx / 256) % 256;
    int bc = idx / 65536;
    float acc = 0.f;
#pragma unroll
    for (int t = 0; t < 6; ++t) {
        int u = x - 1 + t;
        if (u >= 0 && u <= 258 && (u & 1) == 0) {
            int ix = u >> 1;
            size_t o = ((size_t)bc * 256 + y) * 130 + ix;
            acc += c_dec_lo[t] * loW[o] + c_dec_hi[t] * hiW[o];
        }
    }
    q[idx] = acc;
}

// ------------------------- conv1x1 (128x64) via WMMA ------------------------
// grid (1024, B), block 128 (4 wave32). Each block: one batch, 64 columns.
// x tile staged by the Tensor Data Mover, repacked f32->f16 fragment-major.
__global__ __launch_bounds__(128)
void conv1x1_wmma_kernel(const float* __restrict__ x, const float* __restrict__ W,
                         float* __restrict__ y) {
    __shared__ alignas(16) _Float16 Wl[128 * 64];     // 16 KB
    __shared__ alignas(16) float    XlF[64 * 16];     // 4 KB (TDM destination)
    __shared__ alignas(32) _Float16 Xf[2 * 32 * 16];  // 2 KB fragment-major
    int b  = blockIdx.y;
    int p0 = blockIdx.x * 64;
    int t  = threadIdx.x;
    int lane = t & 31, wave = t >> 5;

#pragma unroll
    for (int i = 0; i < 64; ++i) Wl[t + i * 128] = (_Float16)W[t + i * 128];
    __syncthreads();

    v16h a[2][2];
#pragma unroll
    for (int mi = 0; mi < 2; ++mi) {
        int mbase = (wave * 2 + mi) * 16;
        a[mi][0] = frag_A_rowmajor(Wl, 64, mbase, 0,  lane);
        a[mi][1] = frag_A_rowmajor(Wl, 64, mbase, 32, lane);
    }

    const float* xb = x + (size_t)b * 64  * HWFULL;
    float*       yb = y + (size_t)b * 128 * HWFULL;

    for (int nt = 0; nt < 4; ++nt) {
        int p = p0 + nt * 16;
        __syncthreads();                 // previous tile fully consumed
#ifdef HAVE_TDM
        if (t < 32) {                    // one wave drives the TDM
            tdm_load_2d_f32((unsigned)(uintptr_t)XlF,
                            (unsigned long long)(uintptr_t)(xb + p),
                            /*tile_d0=*/16, /*tile_d1=*/64,
                            /*tensor_d0=*/HWFULL, /*tensor_d1=*/64,
                            /*stride0=*/HWFULL);
            wait_tensorcnt0();
        }
#else
#pragma unroll
        for (int i = 0; i < 8; ++i) {
            int e = t + i * 128;
            XlF[e] = xb[(size_t)(e >> 4) * HWFULL + p + (e & 15)];
        }
#endif
        __syncthreads();                 // XlF ready
        // repack f32 [k][n] -> f16 fragment-major (pairs share a lane-half)
#pragma unroll
        for (int i = 0; i < 4; ++i) {
            int e  = t + i * 128;        // 512 pairs
            int kp = e >> 4;             // 0..31
            int n  = e & 15;
            int k0 = kp * 2;
            v2h pr;
            pr[0] = (_Float16)XlF[k0 * 16 + n];
            pr[1] = (_Float16)XlF[(k0 + 1) * 16 + n];
            int kc = k0 >> 5, kk = k0 & 31;
            int lf = ((kk >= 16) ? 16 : 0) + n;
            *(v2h*)&Xf[((kc * 32 + lf) << 4) + (kk & 15)] = pr;
        }
        __syncthreads();                 // Xf ready
        v16h b0 = frag_B_fragmajor(Xf, 0,  lane);
        v16h b1 = frag_B_fragmajor(Xf, 32, lane);
#pragma unroll
        for (int mi = 0; mi < 2; ++mi) {
            v8f c = {};
            c = wmma_f16f32(a[mi][0], b0, c);
            c = wmma_f16f32(a[mi][1], b1, c);
            int mbase = (wave * 2 + mi) * 16;
            int pp = p + (lane & 15);
#pragma unroll
            for (int r = 0; r < 8; ++r) {
                int m = mbase + r + ((lane >= 16) ? 8 : 0);
                yb[(size_t)m * HWFULL + pp] = c[r];
            }
        }
    }
}

// ----------------------- attention statistics (split-K) ---------------------
__global__ void zero_kernel(float* p, int n) {
    int i = blockIdx.x * blockDim.x + threadIdx.x;
    if (i < n) p[i] = 0.f;
}

__device__ inline void red_atomic(float v, float* addr) {
#pragma unroll
    for (int off = 16; off > 0; off >>= 1) v += __shfl_down(v, off, 32);
    if ((threadIdx.x & 31) == 0) atomicAdd(addr, v);
}

// stats per (b,h) [152]: [0:8]=sum q^2, [8:16]=sum k_r^2, [16:24]=sum k_d^2,
// [24:88]=G1[c][d], [88:152]=G2[c][d]
__global__ __launch_bounds__(256)
void stats_kernel(const float* __restrict__ q, const float* __restrict__ kvr,
                  const float* __restrict__ kvd, float* __restrict__ stats) {
    int bh = blockIdx.y;
    int b = bh >> 3, h = bh & 7;
    size_t qbase = ((size_t)b * 64  + h * 8) * HWFULL;
    size_t kbase = ((size_t)b * 128 + h * 8) * HWFULL;  // k = channels 0..63 of kv
    float qq[8] = {0}, kk1[8] = {0}, kk2[8] = {0};
    float g1[8][8] = {{0}}, g2[8][8] = {{0}};

    for (int i = 0; i < 8; ++i) {
        int n = blockIdx.x * 2048 + i * 256 + threadIdx.x;
        float qv[8], k1v[8], k2v[8];
        __builtin_prefetch(q   + qbase + n + 2048, 0, 1);   // global_prefetch_b8
        __builtin_prefetch(kvr + kbase + n + 2048, 0, 1);
#pragma unroll
        for (int c = 0; c < 8; ++c) {
            qv[c]  = q  [qbase + (size_t)c * HWFULL + n];
            k1v[c] = kvr[kbase + (size_t)c * HWFULL + n];
            k2v[c] = kvd[kbase + (size_t)c * HWFULL + n];
        }
#pragma unroll
        for (int c = 0; c < 8; ++c) {
            qq[c]  += qv[c]  * qv[c];
            kk1[c] += k1v[c] * k1v[c];
            kk2[c] += k2v[c] * k2v[c];
#pragma unroll
            for (int d = 0; d < 8; ++d) {
                g1[c][d] += qv[c] * k1v[d];
                g2[c][d] += qv[c] * k2v[d];
            }
        }
    }
    float* sb = stats + bh * 152;
#pragma unroll
    for (int c = 0; c < 8; ++c) {
        red_atomic(qq[c],  sb + c);
        red_atomic(kk1[c], sb + 8 + c);
        red_atomic(kk2[c], sb + 16 + c);
#pragma unroll
        for (int d = 0; d < 8; ++d) {
            red_atomic(g1[c][d], sb + 24 + c * 8 + d);
            red_atomic(g2[c][d], sb + 88 + c * 8 + d);
        }
    }
}

__global__ void softmax_kernel(const float* __restrict__ stats, const float* __restrict__ temp,
                               float* __restrict__ attn1, float* __restrict__ attn2) {
    int bh = blockIdx.x;
    int h  = bh & 7;
    const float* sb = stats + bh * 152;
    int t = threadIdx.x;
    int c = t >> 3, d = t & 7;
    __shared__ float l1[64], l2[64];
    float qn  = fmaxf(sqrtf(sb[c]),      1e-12f);
    float k1n = fmaxf(sqrtf(sb[8 + d]),  1e-12f);
    float k2n = fmaxf(sqrtf(sb[16 + d]), 1e-12f);
    float T = temp[h];
    l1[t] = sb[24 + c * 8 + d] / (qn * k1n) * T;   // attn1: q and k normalized
    l2[t] = sb[88 + c * 8 + d] / k2n * T;          // attn2: only k normalized
    __syncthreads();
    float m1 = -1e30f, m2 = -1e30f;
#pragma unroll
    for (int j = 0; j < 8; ++j) { m1 = fmaxf(m1, l1[c * 8 + j]); m2 = fmaxf(m2, l2[c * 8 + j]); }
    float s1 = 0.f, s2 = 0.f;
#pragma unroll
    for (int j = 0; j < 8; ++j) { s1 += expf(l1[c * 8 + j] - m1); s2 += expf(l2[c * 8 + j] - m2); }
    attn1[bh * 64 + t] = expf(l1[t] - m1) / s1;
    attn2[bh * 64 + t] = expf(l2[t] - m2) / s2;
}

// --------- fused: (attn1@v_r) * (attn2@v_d) then conv1x1(w_proj) via WMMA ---
// grid (4096, B), block 128; product tile written fragment-major in LDS.
__global__ __launch_bounds__(128)
void fused_attn_proj_kernel(const float* __restrict__ kvr, const float* __restrict__ kvd,
                            const float* __restrict__ attn1, const float* __restrict__ attn2,
                            const float* __restrict__ Wp, float* __restrict__ out) {
    __shared__ float    a1l[512], a2l[512];
    __shared__ alignas(16) _Float16 wpl[64 * 64];      // 8 KB
    __shared__ alignas(32) _Float16 prodF[2 * 32 * 16];// 2 KB fragment-major
    int b  = blockIdx.y;
    int p0 = blockIdx.x * 16;
    int t  = threadIdx.x;

#pragma unroll
    for (int i = 0; i < 4; ++i) {
        a1l[t + i * 128] = attn1[b * 512 + t + i * 128];
        a2l[t + i * 128] = attn2[b * 512 + t + i * 128];
    }
#pragma unroll
    for (int i = 0; i < 32; ++i) wpl[t + i * 128] = (_Float16)Wp[t + i * 128];

    // thread -> (head h = t/16, column n = t&15); v channels = 64 + h*8 + d
    int n = t & 15, h = t >> 4;
    const float* vr = kvr + ((size_t)b * 128 + 64 + h * 8) * HWFULL + p0 + n;
    const float* vd = kvd + ((size_t)b * 128 + 64 + h * 8) * HWFULL + p0 + n;
    float vrv[8], vdv[8];
#pragma unroll
    for (int d = 0; d < 8; ++d) { vrv[d] = vr[(size_t)d * HWFULL]; vdv[d] = vd[(size_t)d * HWFULL]; }
    __syncthreads();

    // cin = h*8+cl; all 8 cl stay in one lane-half -> one v8h store
    v8h vals;
#pragma unroll
    for (int cl = 0; cl < 8; ++cl) {
        float s = 0.f, u = 0.f;
#pragma unroll
        for (int d = 0; d < 8; ++d) {
            s += a1l[h * 64 + cl * 8 + d] * vrv[d];
            u += a2l[h * 64 + cl * 8 + d] * vdv[d];
        }
        vals[cl] = (_Float16)(s * u);
    }
    {
        int kcb = h >> 2;                         // K-chunk (0: cin<32, 1: cin>=32)
        int lf  = ((h & 2) ? 16 : 0) + n;         // lane within fragment
        int jb  = (h & 1) * 8;                    // element offset
        *(v8h*)&prodF[(((kcb * 32) + lf) << 4) + jb] = vals;
    }
    __syncthreads();

    int lane = t & 31, wave = t >> 5;
    int mbase = wave * 16;                        // 4 waves x 16 rows = 64 out ch
    v16h aa0 = frag_A_rowmajor(wpl, 64, mbase, 0,  lane);
    v16h aa1 = frag_A_rowmajor(wpl, 64, mbase, 32, lane);
    v16h bb0 = frag_B_fragmajor(prodF, 0,  lane);
    v16h bb1 = frag_B_fragmajor(prodF, 32, lane);
    v8f c = {};
    c = wmma_f16f32(aa0, bb0, c);
    c = wmma_f16f32(aa1, bb1, c);

    float* ob = out + (size_t)b * 64 * HWFULL;
    int pp = p0 + (lane & 15);
#pragma unroll
    for (int r = 0; r < 8; ++r) {
        int m = mbase + r + ((lane >= 16) ? 8 : 0);
        ob[(size_t)m * HWFULL + pp] = c[r];
    }
}

// --------------------------------- launch -----------------------------------

extern "C" void kernel_launch(void* const* d_in, const int* in_sizes, int n_in,
                              void* d_out, int out_size, void* d_ws, size_t ws_size,
                              hipStream_t stream) {
    (void)in_sizes; (void)n_in; (void)out_size; (void)ws_size;
    const float* rgb   = (const float*)d_in[0];
    const float* depth = (const float*)d_in[1];
    const float* temp  = (const float*)d_in[2];
    const float* Wqr   = (const float*)d_in[3];
    const float* Wqd   = (const float*)d_in[4];
    const float* Wqc   = (const float*)d_in[5];
    const float* Wm1   = (const float*)d_in[6];
    const float* Wm5   = (const float*)d_in[7];
    const float* Wm7   = (const float*)d_in[8];
    const float* Wm9   = (const float*)d_in[9];
    const float* Wproj = (const float*)d_in[10];
    float* out = (float*)d_out;
    float* ws  = (float*)d_ws;

    const size_t S_Q = 16777216, S_KV = 33554432, S_ROW = 8519680, S_SB = 4326400;
    float* q     = ws;
    float* kvr   = ws + S_Q;
    float* kvd   = kvr + S_KV;
    float* lin   = kvd + S_KV;          // 33.55M scratch
    float* m1    = lin + S_KV;          // 17.31M scratch
    float* stats = m1 + 4 * S_SB;
    float* a1    = stats + 32 * 152;
    float* a2    = a1 + 32 * 64;

    const int T_ROW = NB * 64 * 256 * 130;   // 8,519,680
    const int T_SB  = NB * 64 * 130 * 130;   // 4,326,400
    const int T_Q   = NB * 64 * 256 * 256;   // 16,777,216
    const int T_KV  = NB * 128 * 256 * 256;  // 33,554,432

    // ---- MFE: q = idwt2(dw3(dwt2(rgb))) ----
    float* tmpLo = lin;            float* tmpHi = lin + S_ROW;
    dwt_row_kernel<<<(T_ROW + 255) / 256, 256, 0, stream>>>(rgb, tmpLo, tmpHi, T_ROW);
    float* ll = m1; float* lh = m1 + S_SB; float* hl = m1 + 2 * S_SB; float* hh = m1 + 3 * S_SB;
    dwt_col_kernel<<<(T_SB + 255) / 256, 256, 0, stream>>>(tmpLo, ll, lh, T_SB);
    dwt_col_kernel<<<(T_SB + 255) / 256, 256, 0, stream>>>(tmpHi, hl, hh, T_SB);
    float* fll = lin; float* flh = lin + S_SB; float* fhl = lin + 2 * S_SB; float* fhh = lin + 3 * S_SB;
    dw3_kernel<<<(T_SB + 255) / 256, 256, 0, stream>>>(ll, Wm1, fll, 64, 130, 130, T_SB);
    dw3_kernel<<<(T_SB + 255) / 256, 256, 0, stream>>>(lh, Wm5, flh, 64, 130, 130, T_SB);
    dw3_kernel<<<(T_SB + 255) / 256, 256, 0, stream>>>(hl, Wm7, fhl, 64, 130, 130, T_SB);
    dw3_kernel<<<(T_SB + 255) / 256, 256, 0, stream>>>(hh, Wm9, fhh, 64, 130, 130, T_SB);
    float* loW = m1; float* hiW = m1 + S_ROW;
    idwt_col_kernel<<<(T_ROW + 255) / 256, 256, 0, stream>>>(fll, flh, loW, T_ROW);
    idwt_col_kernel<<<(T_ROW + 255) / 256, 256, 0, stream>>>(fhl, fhh, hiW, T_ROW);
    idwt_row_kernel<<<(T_Q + 255) / 256, 256, 0, stream>>>(loW, hiW, q, T_Q);

    // ---- kv = dw3(conv1x1(x)) for rgb and depth (WMMA GEMMs, TDM staging) ----
    conv1x1_wmma_kernel<<<dim3(1024, NB), 128, 0, stream>>>(rgb, Wqr, lin);
    dw3_kernel<<<(T_KV + 255) / 256, 256, 0, stream>>>(lin, Wqc, kvr, 128, 256, 256, T_KV);
    conv1x1_wmma_kernel<<<dim3(1024, NB), 128, 0, stream>>>(depth, Wqd, lin);
    dw3_kernel<<<(T_KV + 255) / 256, 256, 0, stream>>>(lin, Wqc, kvd, 128, 256, 256, T_KV);

    // ---- attention statistics -> softmax ----
    zero_kernel<<<(32 * 152 + 255) / 256, 256, 0, stream>>>(stats, 32 * 152);
    stats_kernel<<<dim3(32, 32), 256, 0, stream>>>(q, kvr, kvd, stats);
    softmax_kernel<<<32, 64, 0, stream>>>(stats, temp, a1, a2);

    // ---- fused attention-apply + projection (WMMA) ----
    fused_attn_proj_kernel<<<dim3(4096, NB), 128, 0, stream>>>(kvr, kvd, a1, a2, Wproj, out);
}
